// ResCNN_ASP_SpeakerEncoder_13632226197641
// MI455X (gfx1250) — matrix operand order repaired and checked by reading, not compiled, not run on hardware
//
#include <hip/hip_runtime.h>
#include <hip/hip_bf16.h>

typedef __attribute__((ext_vector_type(16))) _Float16 v16h;
typedef __attribute__((ext_vector_type(8)))  _Float16 v8h;
typedef __attribute__((ext_vector_type(8)))  float    v8f;

#define TLEN  3000
#define BATCH 64
#define NBINS 257
#define NFILT 80

// ---------------------------------------------------------------- helpers ---

static __device__ __forceinline__ v8f zero8() {
  v8f z;
#pragma unroll
  for (int i = 0; i < 8; ++i) z[i] = 0.0f;
  return z;
}

static __device__ __forceinline__ v16h cat8(v8h lo, v8h hi) {
  return __builtin_shufflevector(lo, hi, 0, 1, 2, 3, 4, 5, 6, 7,
                                 8, 9, 10, 11, 12, 13, 14, 15);
}

// A fragment: 16(M) x 32(K) f16, ROW-major with stride ld (ld % 8 == 0, base
// 16B aligned).  Lane layout (ISA 7.12.2): lanes 0-15: M=lane, halves
// {0..7}=K0..7,{8..15}=K16..23; lanes 16-31: M=lane-16, K{8..15, 24..31}.
// Each half-run is 8 contiguous halves -> two 16B loads.
static __device__ __forceinline__ v16h load_a16(const _Float16* p, int ld) {
  const int lane = threadIdx.x & 31;
  const int m    = lane & 15;
  const int kb   = (lane < 16) ? 0 : 8;
  const _Float16* row = p + m * ld + kb;
  const v8h lo = *(const v8h*)(row);
  const v8h hi = *(const v8h*)(row + 16);
  return cat8(lo, hi);
}

// B fragment: 32(K) x 16(N) f16 stored COLUMN-major: p[n*ldk + k]
// (ldk % 8 == 0, base 16B aligned).  lanes 0-15: N=lane, K{0..7,16..23};
// lanes 16-31: N=lane-16, K{8..15,24..31}.  Two 8-half runs -> two 16B loads.
static __device__ __forceinline__ v16h load_b16cm(const _Float16* p, int ldk) {
  const int lane = threadIdx.x & 31;
  const int n    = lane & 15;
  const int kb   = (lane < 16) ? 0 : 8;
  const _Float16* col = p + n * ldk + kb;
  const v8h lo = *(const v8h*)(col);
  const v8h hi = *(const v8h*)(col + 16);
  return cat8(lo, hi);
}

static __device__ __forceinline__ v8f wmma16(v16h a, v16h b, v8f c) {
  return __builtin_amdgcn_wmma_f32_16x16x32_f16(false, a, false, b, (short)0, c,
                                                false, false);
}

// ------------------------------------------------- 1) fused filterbank+W1 ---
// Mt[o][f] = sum_n fb_mod[n,f] * w_lin1[o,n]  (transposed store, f padded to
// 288 so B-fragments are 16B-aligned contiguous K-runs in global memory).
__global__ void k_build_M(const float* __restrict__ bp,
                          const float* __restrict__ w1,
                          _Float16* __restrict__ Mt) {
  __shared__ float sb[NFILT + 2];
  __shared__ float srt[NFILT + 2];
  const int tid = threadIdx.x;
  if (tid < NFILT + 2) sb[tid] = bp[tid];
  __syncthreads();
  if (tid < NFILT + 2) {
    float v = sb[tid];
    int rank = 0;
    for (int j = 0; j < NFILT + 2; ++j) {
      float u = sb[j];
      rank += (u < v) || (u == v && j < tid);
    }
    srt[rank] = v;
  }
  __syncthreads();
  for (int idx = tid; idx < 288 * 64; idx += blockDim.x) {
    const int f = idx >> 6;
    const int o = idx & 63;
    float acc = 0.0f;
    if (f < NBINS) {
      if (f == 0) acc += w1[o * NFILT + 0];          // override row: e0
      for (int n = 1; n < NFILT - 1; ++n) {          // rows 1..78 triangular
        const float b0 = srt[n], b1v = srt[n + 1], b2v = srt[n + 2];
        const int i0  = (int)floorf(b0);
        const int i1i = (int)floorf(b1v);
        const int i2i = (int)floorf(b2v);
        float fb = 0.0f;
        if (f >= i0 && f < i1i) {
          float d = (b1v - b0) * (b1v - b0);
          fb = ((float)f - b0) / (d > 0.0f ? d : 1.0f);
        } else if (f >= i1i && f < i2i) {
          float d = (b2v - b1v) * (b2v - b1v);
          fb = (b2v - (float)f) / (d > 0.0f ? d : 1.0f);
        }
        acc += fb * w1[o * NFILT + n];
      }
    }
    Mt[o * 288 + f] = (_Float16)acc;
  }
}

// ------------------------------------- 2) h = relu(x @ M + b1), WMMA GEMM ---
// rows = B*T = 192000, K = 288, N = 64.  B-fragments straight from global Mt
// (36 KB, L2-resident).  Output h f16 [B][64][T].
__global__ void k_gemm1(const float* __restrict__ x,
                        const _Float16* __restrict__ Mt,
                        const float* __restrict__ b1,
                        _Float16* __restrict__ h) {
  __shared__ __attribute__((aligned(16))) _Float16 Xs[128][40];
  const int tid  = threadIdx.x;
  const int wave = tid >> 5;
  const int lane = tid & 31;
  const int rowbase = blockIdx.x * 128;
  v8f acc[4];
#pragma unroll
  for (int j = 0; j < 4; ++j) acc[j] = zero8();

  for (int ks = 0; ks < 9; ++ks) {
    __syncthreads();
    // stage X: 128 rows x 32 k, one aligned v8h store per chunk
#pragma unroll
    for (int c2 = 0; c2 < 2; ++c2) {
      const int c    = tid + 256 * c2;       // 0..511
      const int r    = c >> 2;
      const int koff = (c & 3) * 8;
      v8h v;
#pragma unroll
      for (int j = 0; j < 8; ++j) {
        const int f = ks * 32 + koff + j;
        v[j] = (_Float16)((f < NBINS)
                              ? x[(long long)(rowbase + r) * NBINS + f]
                              : 0.0f);
      }
      *(v8h*)(&Xs[r][koff]) = v;
    }
    __syncthreads();
    const v16h a = load_a16(&Xs[wave * 16][0], 40);
#pragma unroll
    for (int jt = 0; jt < 4; ++jt) {
      const v16h bf = load_b16cm(Mt + (jt * 16) * 288 + ks * 32, 288);
      acc[jt] = wmma16(a, bf, acc[jt]);
    }
  }
  const int n    = lane & 15;
  const int moff = (lane < 16) ? 0 : 8;
#pragma unroll
  for (int jt = 0; jt < 4; ++jt) {
    const int o = jt * 16 + n;
    const float bia = b1[o];
#pragma unroll
    for (int r = 0; r < 8; ++r) {
      const int row = rowbase + wave * 16 + moff + r;
      float v = acc[jt][r] + bia;
      v = v > 0.0f ? v : 0.0f;
      const int bb = row / TLEN;
      const int t  = row - bb * TLEN;
      h[((long long)bb * 64 + o) * TLEN + t] = (_Float16)v;
    }
  }
}

// ------------------------- 3) dilated conv (3 taps) as accumulated WMMA ----
// Input tile staged COLUMN-major tileT[col][ci] (stride CI+8) so B-fragments
// are two aligned ds_load_b128 per lane; taps select the column window.
// mode 0: out = raw ; mode 1: out = raw+r1, out2 = raw+2*r1 ; mode 2: raw+r1.
template <int CI, int DIL>
__global__ void k_dconv(const _Float16* __restrict__ in,
                        const float* __restrict__ w,
                        const float* __restrict__ bias,
                        const _Float16* __restrict__ r1,
                        _Float16* __restrict__ out,
                        _Float16* __restrict__ out2, int mode) {
  constexpr int WIN = 16 + 2 * DIL;
  constexpr int KC  = CI / 32;
  constexpr int CIP = CI + 8;                 // pad: 16B multiple, bank-skewed
  __shared__ __attribute__((aligned(16))) _Float16 tileT[WIN][CIP];
  const int tid  = threadIdx.x;
  const int wave = tid >> 5;
  const int lane = tid & 31;
  const int bb   = blockIdx.y;
  const int co0  = wave * 16;
  const int m    = lane & 15;
  const int kb   = (lane < 16) ? 0 : 8;

  v16h afr[3][KC];                            // per-wave resident weight frags
#pragma unroll
  for (int tap = 0; tap < 3; ++tap)
#pragma unroll
    for (int kc = 0; kc < KC; ++kc) {
      v16h a;
#pragma unroll
      for (int i = 0; i < 8; ++i) {
        const int k0 = kc * 32 + kb + i;
        a[i]     = (_Float16)w[((co0 + m) * CI + k0) * 3 + tap];
        a[8 + i] = (_Float16)w[((co0 + m) * CI + k0 + 16) * 3 + tap];
      }
      afr[tap][kc] = a;
    }

  constexpr int NT = (TLEN + 15) / 16;        // 188 t-tiles
  int tend = blockIdx.x * 8 + 8;
  if (tend > NT) tend = NT;
  for (int tt = blockIdx.x * 8; tt < tend; ++tt) {
    const int t0 = tt * 16;
    if (tid < CI) {                            // prefetch next window
      int tn = t0 + 16 - DIL;
      if (tn >= TLEN) tn = TLEN - 1;
      __builtin_prefetch(&in[((long long)bb * CI + tid) * TLEN + tn], 0, 1);
    }
    __syncthreads();
    for (int e = tid; e < CI * WIN; e += 256) {
      const int ci = e / WIN, col = e - ci * WIN;   // coalesced along t
      const int ts = t0 - DIL + col;
      _Float16 v = (_Float16)0.0f;
      if (ts >= 0 && ts < TLEN) v = in[((long long)bb * CI + ci) * TLEN + ts];
      tileT[col][ci] = v;
    }
    __syncthreads();
    v8f acc = zero8();
#pragma unroll
    for (int tap = 0; tap < 3; ++tap)
#pragma unroll
      for (int kc = 0; kc < KC; ++kc) {
        const v16h bf = load_b16cm(&tileT[tap * DIL][kc * 32], CIP);
        acc = wmma16(afr[tap][kc], bf, acc);
      }
    const int n    = lane & 15;
    const int moff = (lane < 16) ? 0 : 8;
    const int t    = t0 + n;
    if (t < TLEN) {
#pragma unroll
      for (int r = 0; r < 8; ++r) {
        const int co = co0 + moff + r;
        const float raw = acc[r] + bias[co];
        const long long idx = ((long long)bb * 128 + co) * TLEN + t;
        if (mode == 0) {
          out[idx] = (_Float16)raw;
        } else if (mode == 1) {
          const float v1 = (float)r1[idx];
          out[idx]  = (_Float16)(raw + v1);
          out2[idx] = (_Float16)(raw + 2.0f * v1);
        } else {
          const float v1 = (float)r1[idx];
          out[idx] = (_Float16)(raw + v1);
        }
      }
    }
  }
}

// ------------------------------- 4) channel GEMM (ASP projections), WMMA ---
template <int CI, int CO, int ACT>
__global__ void k_chan_gemm(const _Float16* __restrict__ in,
                            const float* __restrict__ w,
                            const float* __restrict__ bias,
                            _Float16* __restrict__ out) {
  constexpr int NT     = 128 / CO;            // t-tiles per block
  constexpr int COLS   = 16 * NT;
  constexpr int KC     = CI / 32;
  constexpr int CTILES = CO / 16;
  constexpr int CIP    = CI + 8;
  __shared__ __attribute__((aligned(16))) _Float16 tileT[COLS][CIP];
  const int tid  = threadIdx.x;
  const int wave = tid >> 5;
  const int lane = tid & 31;
  const int bb   = blockIdx.y;
  const int wt   = wave / CTILES;
  const int co0  = (wave % CTILES) * 16;
  const int m    = lane & 15;
  const int kb   = (lane < 16) ? 0 : 8;

  v16h afr[KC];
#pragma unroll
  for (int kc = 0; kc < KC; ++kc) {
    v16h a;
#pragma unroll
    for (int i = 0; i < 8; ++i) {
      a[i]     = (_Float16)w[(co0 + m) * CI + kc * 32 + kb + i];
      a[8 + i] = (_Float16)w[(co0 + m) * CI + kc * 32 + kb + 16 + i];
    }
    afr[kc] = a;
  }
  const int tbase = blockIdx.x * COLS;
  for (int e = tid; e < CI * COLS; e += 256) {
    const int ci = e / COLS, col = e - ci * COLS;   // coalesced along t
    const int ts = tbase + col;
    tileT[col][ci] =
        (ts < TLEN) ? in[((long long)bb * CI + ci) * TLEN + ts] : (_Float16)0.0f;
  }
  __syncthreads();
  v8f acc = zero8();
#pragma unroll
  for (int kc = 0; kc < KC; ++kc) {
    const v16h bf = load_b16cm(&tileT[wt * 16][kc * 32], CIP);
    acc = wmma16(afr[kc], bf, acc);
  }
  const int n    = lane & 15;
  const int moff = (lane < 16) ? 0 : 8;
  const int t    = tbase + wt * 16 + n;
  if (t < TLEN) {
#pragma unroll
    for (int r = 0; r < 8; ++r) {
      const int co = co0 + moff + r;
      float v = acc[r] + bias[co];
      if (ACT == 1) v = tanhf(v);
      out[((long long)bb * CO + co) * TLEN + t] = (_Float16)v;
    }
  }
}

// --------------------- 5) softmax over T + weighted mean/std (one wave/row) -
__global__ void k_asp_stats(const _Float16* __restrict__ e,
                            const _Float16* __restrict__ i3,
                            float* __restrict__ pool) {
  const int wave = threadIdx.x >> 5;
  const int lane = threadIdx.x & 31;
  const int row  = blockIdx.x * 8 + wave;     // b*128 + c
  const int bb   = row >> 7;
  const int c    = row & 127;
  const _Float16* ep = e + (long long)row * TLEN;
  const _Float16* vp = i3 + (long long)row * TLEN;
  float mx = -1e30f;
  for (int t = lane; t < TLEN; t += 32) mx = fmaxf(mx, (float)ep[t]);
#pragma unroll
  for (int o = 16; o > 0; o >>= 1) mx = fmaxf(mx, __shfl_xor(mx, o, 32));
  float Z = 0.0f, S1 = 0.0f, S2 = 0.0f;
  for (int t = lane; t < TLEN; t += 32) {
    const float wgt = __expf((float)ep[t] - mx);
    const float v = (float)vp[t];
    Z += wgt; S1 += wgt * v; S2 += wgt * v * v;
  }
#pragma unroll
  for (int o = 16; o > 0; o >>= 1) {
    Z  += __shfl_xor(Z, o, 32);
    S1 += __shfl_xor(S1, o, 32);
    S2 += __shfl_xor(S2, o, 32);
  }
  if (lane == 0) {
    const float mean = S1 / Z;
    const float var = S2 / Z - mean * mean;
    pool[bb * 256 + c]       = mean;
    pool[bb * 256 + 128 + c] = sqrtf(fmaxf(var, 1e-9f));
  }
}

// ------------------------------------ 6) layernorm + final 256->512 linear --
__global__ void k_final(const float* __restrict__ pool,
                        const float* __restrict__ gamma,
                        const float* __restrict__ beta,
                        const float* __restrict__ w2,
                        const float* __restrict__ b2,
                        float* __restrict__ out) {
  __shared__ float sv[256];
  __shared__ float r1[256];
  __shared__ float r2[256];
  const int tid = threadIdx.x;
  const int bb  = blockIdx.x;
  const float v = pool[bb * 256 + tid];
  r1[tid] = v; r2[tid] = v * v;
  __syncthreads();
  for (int st = 128; st > 0; st >>= 1) {
    if (tid < st) { r1[tid] += r1[tid + st]; r2[tid] += r2[tid + st]; }
    __syncthreads();
  }
  const float mu  = r1[0] * (1.0f / 256.0f);
  const float var = r2[0] * (1.0f / 256.0f) - mu * mu;
  sv[tid] = (v - mu) * rsqrtf(var + 1e-5f) * gamma[tid] + beta[tid];
  __syncthreads();
  for (int o = tid; o < 512; o += 256) {
    float acc = b2[o];
    for (int k = 0; k < 256; ++k) acc += sv[k] * w2[o * 256 + k];
    out[bb * 512 + o] = acc;
  }
}

// --------------------------------------------------------------- launcher ---
extern "C" void kernel_launch(void* const* d_in, const int* in_sizes, int n_in,
                              void* d_out, int out_size, void* d_ws, size_t ws_size,
                              hipStream_t stream) {
  (void)in_sizes; (void)n_in; (void)out_size; (void)ws_size;
  const float* x      = (const float*)d_in[0];
  const float* bp     = (const float*)d_in[1];
  const float* w_lin1 = (const float*)d_in[2];
  const float* b_lin1 = (const float*)d_in[3];
  const float* w_c1   = (const float*)d_in[4];
  const float* b_c1   = (const float*)d_in[5];
  const float* w_c2   = (const float*)d_in[6];
  const float* b_c2   = (const float*)d_in[7];
  const float* w_c3   = (const float*)d_in[8];
  const float* b_c3   = (const float*)d_in[9];
  const float* w_a1   = (const float*)d_in[10];
  const float* b_a1   = (const float*)d_in[11];
  const float* w_a2   = (const float*)d_in[12];
  const float* b_a2   = (const float*)d_in[13];
  const float* gamma  = (const float*)d_in[14];
  const float* beta   = (const float*)d_in[15];
  const float* w_l2   = (const float*)d_in[16];
  const float* b_l2   = (const float*)d_in[17];

  char* ws = (char*)d_ws;
  const long long SZ_H = (long long)BATCH * 64 * TLEN * 2;    // f16
  const long long SZ_I = (long long)BATCH * 128 * TLEN * 2;   // f16
  _Float16* Mt = (_Float16*)(ws);
  _Float16* h  = (_Float16*)(ws + 65536);
  _Float16* i1 = (_Float16*)(ws + 65536 + SZ_H);
  _Float16* i2 = (_Float16*)(ws + 65536 + SZ_H + SZ_I);
  _Float16* s  = (_Float16*)(ws + 65536 + SZ_H + 2 * SZ_I);
  _Float16* i3 = (_Float16*)(ws + 65536 + SZ_H + 3 * SZ_I);
  float* pool  = (float*)(ws + 65536 + SZ_H + 4 * SZ_I);
  _Float16* a  = h;    // h dead after conv1
  _Float16* e  = i1;   // i1 dead after conv2

  k_build_M<<<1, 256, 0, stream>>>(bp, w_lin1, Mt);
  k_gemm1<<<1500, 256, 0, stream>>>(x, Mt, b_lin1, h);
  dim3 cgrid(24, BATCH);
  k_dconv<64, 2><<<cgrid, 256, 0, stream>>>(h, w_c1, b_c1, nullptr, i1, nullptr, 0);
  k_dconv<128, 3><<<cgrid, 256, 0, stream>>>(i1, w_c2, b_c2, i1, i2, s, 1);
  k_dconv<128, 4><<<cgrid, 256, 0, stream>>>(s, w_c3, b_c3, s, i3, nullptr, 2);
  k_chan_gemm<128, 64, 1><<<dim3(94, BATCH), 256, 0, stream>>>(i3, w_a1, b_a1, a);
  k_chan_gemm<64, 128, 0><<<dim3(188, BATCH), 256, 0, stream>>>(a, w_a2, b_a2, e);
  k_asp_stats<<<1024, 256, 0, stream>>>(e, i3, pool);
  k_final<<<64, 256, 0, stream>>>(pool, gamma, beta, w_l2, b_l2, (float*)d_out);
}
